// MultiTaskNet_M_44470091383233
// MI455X (gfx1250) — matrix operand order repaired and checked by reading, not compiled, not run on hardware
//
#include <hip/hip_runtime.h>
#include <hip/hip_bf16.h>
#include <math.h>

typedef __attribute__((ext_vector_type(16))) __bf16 v16bf;
typedef __attribute__((ext_vector_type(8)))  __bf16 v8bf;
typedef __attribute__((ext_vector_type(8)))  float  v8f;

#define GRU_H  200
#define GRU_T  1000
#define NTILES 38   // 600 cols padded to 608 = 38 * 16
#define KCH    7    // K=200 padded to 224 = 7 * 32

__device__ __forceinline__ __bf16 f2bf(float f) {
  unsigned u = __builtin_bit_cast(unsigned, f);
  unsigned r = u + 0x7FFFu + ((u >> 16) & 1u);       // round-to-nearest-even
  unsigned short h = (unsigned short)(r >> 16);
  return __builtin_bit_cast(__bf16, h);
}

// ---------------------------------------------------------------------------
// Convert f32 [M x K] row-major -> bf16 [M x Kp] row-major, zero-padded.
// Kp must be a multiple of 8. One thread per 8-element (16B) output chunk.
// ---------------------------------------------------------------------------
__global__ __launch_bounds__(256)
void pad_bf16(const float* __restrict__ src, int K, int Kp, long long M,
              __bf16* __restrict__ dst)
{
  long long idx = (long long)blockIdx.x * blockDim.x + threadIdx.x;
  int cpr = Kp >> 3;                       // chunks per row
  if (idx >= M * cpr) return;
  long long row = idx / cpr;
  int kb = (int)(idx - row * cpr) << 3;

  v8bf o;
  if (kb + 8 <= K) {                       // fully in-range: vector loads
    const float* p = src + row * K + kb;
    float4 f0 = *(const float4*)(p);
    float4 f1 = *(const float4*)(p + 4);
    o[0] = f2bf(f0.x); o[1] = f2bf(f0.y); o[2] = f2bf(f0.z); o[3] = f2bf(f0.w);
    o[4] = f2bf(f1.x); o[5] = f2bf(f1.y); o[6] = f2bf(f1.z); o[7] = f2bf(f1.w);
  } else if (kb >= K) {                    // pure padding
#pragma unroll
    for (int e = 0; e < 8; ++e) o[e] = f2bf(0.f);
  } else {                                 // boundary chunk (rare)
#pragma unroll
    for (int e = 0; e < 8; ++e) {
      int k = kb + e;
      int kc = (k < K) ? k : 0;
      float v = src[row * K + kc];
      o[e] = f2bf((k < K) ? v : 0.f);
    }
  }
  *(v8bf*)(dst + idx * 8) = o;
}

// ---------------------------------------------------------------------------
// C[M,N] = A[M,Kp] @ W[N,Kp]^T + bias.  A,W bf16 (pre-padded, Kp = 32*KC).
// M must be a multiple of 16. One wave per 16x16 output tile.
// Inner loop: 2x16B + 1x32B bf16 loads -> one WMMA. No masks, no converts.
// ---------------------------------------------------------------------------
__global__ __launch_bounds__(256)
void wmma_gemm_bf16(const __bf16* __restrict__ A,
                    const __bf16* __restrict__ W,
                    const float* __restrict__ bias,
                    float* __restrict__ C, int ldc,
                    int M, int N, int KC, int doRelu)
{
  const int lane = threadIdx.x & 31;
  const int wid  = (int)((blockIdx.x * blockDim.x + threadIdx.x) >> 5);
  const int NT = (N + 15) >> 4;
  const int MT = M >> 4;
  if (wid >= MT * NT) return;
  const int mt = wid / NT;
  const int nt = wid - mt * NT;

  const int laneMod = lane & 15;
  const int laneHi  = (lane >> 4) & 1;
  const int Kp    = KC << 5;
  const int arow  = mt * 16 + laneMod;
  const int bcol  = nt * 16 + laneMod;
  const int bclmp = (bcol < N) ? bcol : (N - 1);   // junk cols never stored

  const __bf16* __restrict__ pa = A + (long long)arow  * Kp + laneHi * 16;
  const __bf16* __restrict__ pb = W + (long long)bclmp * Kp + laneHi * 16;

  v8f acc = {0.f,0.f,0.f,0.f,0.f,0.f,0.f,0.f};
  for (int kc = 0; kc < KC; ++kc, pa += 32, pb += 32) {
    v8bf a0 = *(const v8bf*)(pa);        // A-frag elems 0..7
    v8bf a1 = *(const v8bf*)(pa + 8);    // A-frag elems 8..15
    v16bf a = __builtin_shufflevector(a0, a1,
        0,1,2,3,4,5,6,7,8,9,10,11,12,13,14,15);
    v16bf b = *(const v16bf*)(pb);       // B-frag elems 0..15
    acc = __builtin_amdgcn_wmma_f32_16x16x32_bf16(false, a, false, b,
                                                  (short)0, acc, false, false);
  }

  if (bcol < N) {
    float bv = bias ? bias[bcol] : 0.f;
#pragma unroll
    for (int r = 0; r < 8; ++r) {
      int row = mt * 16 + r + laneHi * 8;          // lanes 16-31 hold M+8
      float v = acc[r] + bv;
      if (doRelu) v = fmaxf(v, 0.f);
      C[(long long)row * ldc + bcol] = v;
    }
  }
}

// ---------------------------------------------------------------------------
// GRU recurrence for 16 batch rows per workgroup. Weights live in LDS as
// pre-swizzled WMMA B-fragments for the entire 1000-step loop.
// gi must already contain  x @ w_ih^T + b_ih.
// ---------------------------------------------------------------------------
__global__ __launch_bounds__(256)
void gru_recur(const float* __restrict__ gi,     // [128][T][600]
               const float* __restrict__ whh,    // [600][200]
               const float* __restrict__ bhh,    // [600]
               float* __restrict__ outp,         // seqMode? [128][T][200] : [128][200]
               int seqMode)
{
  extern __shared__ char smem[];
  __bf16* wfragS = (__bf16*)smem;                  // NTILES*KCH*512 bf16 (266KB)
  __bf16* hfragS = wfragS + NTILES * KCH * 512;    // KCH*512 bf16 (7KB)
  float*  ghS    = (float*)(hfragS + KCH * 512);   // 16*608 f32 (38KB)
  float*  bbS    = ghS + 16 * 608;                 // 600 f32

  const int tid   = threadIdx.x;
  const int lane  = tid & 31;
  const int wv    = tid >> 5;
  const int bbase = blockIdx.x * 16;

  // ---- preload w_hh into B-fragment layout (bf16), zero-padded ----
  for (int idx = tid; idx < NTILES * KCH * 512; idx += 256) {
    int f  = idx >> 9, e = idx & 511;
    int l  = e >> 4,  el = e & 15;
    int vg = el >> 1, hf = el & 1;
    int nt = f / KCH, kc = f - nt * KCH;
    int n  = nt * 16 + (l & 15);
    int k  = kc * 32 + ((l >> 4) & 1) * 8 + (vg >> 2) * 16 + (vg & 3) * 2 + hf;
    float v = (n < 600 && k < GRU_H) ? whh[n * GRU_H + k] : 0.f;
    wfragS[idx] = f2bf(v);
  }
  for (int idx = tid; idx < KCH * 512; idx += 256) hfragS[idx] = f2bf(0.f);
  for (int idx = tid; idx < 600; idx += 256) bbS[idx] = bhh[idx];

  float hreg[13];                 // h state lives in registers (fixed mapping)
#pragma unroll
  for (int s = 0; s < 13; ++s) hreg[s] = 0.f;

  __syncthreads();

  for (int t = 0; t < GRU_T; ++t) {
    // ---- gh = h @ w_hh^T : A-fragments (h) hoisted to registers once per
    //      step; each wave then streams only B-fragments from LDS ----
    v16bf afr[KCH];
#pragma unroll
    for (int kc = 0; kc < KCH; ++kc)
      afr[kc] = *(const v16bf*)(hfragS + kc * 512 + lane * 16);

    for (int nt = wv; nt < NTILES; nt += 8) {
      v8f acc = {0.f,0.f,0.f,0.f,0.f,0.f,0.f,0.f};
#pragma unroll
      for (int kc = 0; kc < KCH; ++kc) {
        v16bf b = *(const v16bf*)(wfragS + (nt * KCH + kc) * 512 + lane * 16);
        acc = __builtin_amdgcn_wmma_f32_16x16x32_bf16(false, afr[kc], false, b,
                                                      (short)0, acc, false, false);
      }
      int col = nt * 16 + (lane & 15);
      int rb  = ((lane >> 4) & 1) * 8;
#pragma unroll
      for (int r = 0; r < 8; ++r) ghS[(rb + r) * 608 + col] = acc[r];
    }
    __syncthreads();

    // ---- gate math + state update; repack h into A-fragments ----
#pragma unroll
    for (int s = 0; s < 13; ++s) {
      int idx = tid + s * 256;
      if (idx < 16 * GRU_H) {
        int row = idx / GRU_H;
        int j   = idx - row * GRU_H;
        int gb  = bbase + row;
        const float* g = gi + ((long long)gb * GRU_T + t) * 600;
        float ir = g[j], iz = g[200 + j], in = g[400 + j];
        if (t + 1 < GRU_T) {                 // overlap next step's gi fetch
          __builtin_prefetch(g + 600 + j, 0, 0);
          __builtin_prefetch(g + 800 + j, 0, 0);
          __builtin_prefetch(g + 1000 + j, 0, 0);
        }
        float hr = ghS[row * 608 + j]       + bbS[j];
        float hz = ghS[row * 608 + 200 + j] + bbS[200 + j];
        float hn = ghS[row * 608 + 400 + j] + bbS[400 + j];
        float rg = 1.f / (1.f + __expf(-(ir + hr)));
        float zg = 1.f / (1.f + __expf(-(iz + hz)));
        float ng = tanhf(in + rg * hn);
        float hnew = (1.f - zg) * ng + zg * hreg[s];
        hreg[s] = hnew;
        // scatter into 16-bit A-fragment layout for next step's WMMA
        int kc = j >> 5, kk = j & 31;
        int r8 = (kk & 15) >> 3;
        int ln = row + (r8 << 4);
        int vg = ((kk >= 16) ? 4 : 0) + ((kk & 7) >> 1);
        int hf = kk & 1;
        hfragS[kc * 512 + ln * 16 + vg * 2 + hf] = f2bf(hnew);
        if (seqMode)             outp[((long long)gb * GRU_T + t) * GRU_H + j] = hnew;
        else if (t == GRU_T - 1) outp[gb * GRU_H + j] = hnew;
      }
    }
    __syncthreads();
  }
}

// ---------------------------------------------------------------------------
// Concatenate the 6 head weight matrices into bf16 Wcat[104][128] (K padded
// 100->128 with zeros) and f32 bcat[104].
// ---------------------------------------------------------------------------
__global__ void build_cat(const float* wJ, const float* wmu, const float* wtm,
                          const float* wvth, const float* wjt, const float* wtt,
                          const float* bJ, const float* bmu, const float* btm,
                          const float* bvth, const float* bjt, const float* btt,
                          __bf16* Wcat, float* bcat)
{
  int idx = blockIdx.x * blockDim.x + threadIdx.x;
  if (idx < 104 * 128) {
    int r = idx >> 7, c = idx & 127;
    float v = 0.f;
    if (c < 100) {
      if (r < 64) v = wJ[r * 100 + c];
      else {
        int s = (r - 64) >> 3, rr = (r - 64) & 7;
        const float* w = (s == 0) ? wmu : (s == 1) ? wtm : (s == 2) ? wvth
                       : (s == 3) ? wjt : wtt;
        v = w[rr * 100 + c];
      }
    }
    Wcat[idx] = f2bf(v);
  }
  if (idx < 104) {
    float v;
    if (idx < 64) v = bJ[idx];
    else {
      int s = (idx - 64) >> 3, rr = (idx - 64) & 7;
      const float* b = (s == 0) ? bmu : (s == 1) ? btm : (s == 2) ? bvth
                     : (s == 3) ? bjt : btt;
      v = b[rr];
    }
    bcat[idx] = v;
  }
}

// Ccat[128][104] -> d_out laid out as J(128*64), mu, tau_m, V_th, J_th, tau_th
__global__ void scatter_out(const float* __restrict__ Ccat, float* __restrict__ out)
{
  int idx = blockIdx.x * blockDim.x + threadIdx.x;
  if (idx >= 128 * 104) return;
  int b = idx / 104, c = idx - b * 104;
  float v = Ccat[idx];
  if (c < 64) out[b * 64 + c] = v;
  else {
    int s = (c - 64) >> 3, o = (c - 64) & 7;
    out[128 * 64 + s * 128 * 8 + b * 8 + o] = v;
  }
}

extern "C" void kernel_launch(void* const* d_in, const int* in_sizes, int n_in,
                              void* d_out, int out_size, void* d_ws, size_t ws_size,
                              hipStream_t stream)
{
  const float* x     = (const float*)d_in[0];
  const float* w_ih0 = (const float*)d_in[1];
  const float* w_hh0 = (const float*)d_in[2];
  const float* b_ih0 = (const float*)d_in[3];
  const float* b_hh0 = (const float*)d_in[4];
  const float* w_ih1 = (const float*)d_in[5];
  const float* w_hh1 = (const float*)d_in[6];
  const float* b_ih1 = (const float*)d_in[7];
  const float* b_hh1 = (const float*)d_in[8];
  const float* fc1_w = (const float*)d_in[9];
  const float* fc1_b = (const float*)d_in[10];
  const float* wJ    = (const float*)d_in[11];
  const float* bJ    = (const float*)d_in[12];
  const float* wmu   = (const float*)d_in[13];
  const float* bmu   = (const float*)d_in[14];
  const float* wtm   = (const float*)d_in[15];
  const float* btm   = (const float*)d_in[16];
  const float* wvth  = (const float*)d_in[17];
  const float* bvth  = (const float*)d_in[18];
  const float* wjt   = (const float*)d_in[19];
  const float* bjt   = (const float*)d_in[20];
  const float* wtt   = (const float*)d_in[21];
  const float* btt   = (const float*)d_in[22];

  char* wsb = (char*)d_ws;
  float*  giBuf = (float*)(wsb);                       // 128*1000*600 f32
  float*  h1seq = (float*)(wsb + 307200000);           // 128*1000*200 f32
  float*  hlast = (float*)(wsb + 409600000);           // 128*200 f32
  float*  fbuf  = (float*)(wsb + 409702400);           // 128*100 f32
  float*  bcat  = (float*)(wsb + 409753600);           // 104 f32 (pad to 512B)
  float*  Ccat  = (float*)(wsb + 409754112);           // 128*104 f32
  __bf16* Apad  = (__bf16*)(wsb + 409807360);          // up to 128000*224 bf16
  __bf16* Wpad  = (__bf16*)(wsb + 467151360);          // up to 600*224 bf16
  __bf16* WcatB = (__bf16*)(wsb + 467420160);          // 104*128 bf16

  auto gemmBlocks = [](int M, int N) {
    int tiles = (M >> 4) * ((N + 15) >> 4);
    return (tiles + 7) / 8;             // 8 waves (256 thr) per block
  };
  auto padBlocks = [](long long M, int Kp) {
    long long ch = M * (Kp >> 3);
    return (int)((ch + 255) / 256);
  };
  size_t smemGru = (size_t)(NTILES * KCH * 512 + KCH * 512) * sizeof(__bf16)
                 + (size_t)(16 * 608 + 600) * sizeof(float);   // 320,864 B

  // ---- layer 0 input projection: gi0 = x @ w_ih0^T + b_ih0 ----
  pad_bf16<<<padBlocks(128000, 32), 256, 0, stream>>>(x, 8, 32, 128000, Apad);
  pad_bf16<<<padBlocks(600, 32), 256, 0, stream>>>(w_ih0, 8, 32, 600, Wpad);
  wmma_gemm_bf16<<<gemmBlocks(128000, 600), 256, 0, stream>>>(
      Apad, Wpad, b_ih0, giBuf, 600, 128000, 600, 1, 0);
  // ---- GRU layer 0 -> full sequence h1 ----
  gru_recur<<<8, 256, smemGru, stream>>>(giBuf, w_hh0, b_hh0, h1seq, 1);
  // ---- layer 1 input projection: gi1 = h1 @ w_ih1^T + b_ih1 ----
  pad_bf16<<<padBlocks(128000, 224), 256, 0, stream>>>(h1seq, 200, 224, 128000, Apad);
  pad_bf16<<<padBlocks(600, 224), 256, 0, stream>>>(w_ih1, 200, 224, 600, Wpad);
  wmma_gemm_bf16<<<gemmBlocks(128000, 600), 256, 0, stream>>>(
      Apad, Wpad, b_ih1, giBuf, 600, 128000, 600, 7, 0);
  // ---- GRU layer 1 -> last hidden only ----
  gru_recur<<<8, 256, smemGru, stream>>>(giBuf, w_hh1, b_hh1, hlast, 0);
  // ---- f = relu(last @ fc1_w^T + fc1_b) ----
  pad_bf16<<<padBlocks(128, 224), 256, 0, stream>>>(hlast, 200, 224, 128, Apad);
  pad_bf16<<<padBlocks(100, 224), 256, 0, stream>>>(fc1_w, 200, 224, 100, Wpad);
  wmma_gemm_bf16<<<gemmBlocks(128, 100), 256, 0, stream>>>(
      Apad, Wpad, fc1_b, fbuf, 100, 128, 100, 7, 1);
  // ---- all 6 heads as one GEMM (M=128, K=100->128, N=104) ----
  build_cat<<<(104 * 128 + 255) / 256, 256, 0, stream>>>(
      wJ, wmu, wtm, wvth, wjt, wtt, bJ, bmu, btm, bvth, bjt, btt, WcatB, bcat);
  pad_bf16<<<padBlocks(128, 128), 256, 0, stream>>>(fbuf, 100, 128, 128, Apad);
  wmma_gemm_bf16<<<gemmBlocks(128, 104), 256, 0, stream>>>(
      Apad, WcatB, bcat, Ccat, 104, 128, 104, 4, 0);
  // ---- scatter into d_out layout ----
  scatter_out<<<(128 * 104 + 255) / 256, 256, 0, stream>>>(Ccat, (float*)d_out);
}
